// CALSTM_1194000908624
// MI455X (gfx1250) — matrix operand order repaired and checked by reading, not compiled, not run on hardware
//
#include <hip/hip_runtime.h>
#include <math.h>

#define DEV __device__ __forceinline__

typedef __attribute__((ext_vector_type(16))) __bf16          v16bf;
typedef __attribute__((ext_vector_type(16))) unsigned short  v16u;
typedef __attribute__((ext_vector_type(8)))  unsigned short  v8us;
typedef __attribute__((ext_vector_type(8)))  float           v8f;

static constexpr int B_  = 64;
static constexpr int L_  = 196;
static constexpr int D_  = 512;
static constexpr int H_  = 512;
static constexpr int E_  = 256;
static constexpr int T_  = 128;
static constexpr int ML  = B_ * L_;         // 12544 rows (multiple of 32)
static constexpr int BT  = B_ * T_;         // 8192
static constexpr int G4H = 4 * H_;          // 2048
static constexpr int OUTW = H_ + D_ + E_;   // 1280

// ---------- scalar helpers ----------
DEV unsigned short f2bf(float f) {
  union { float f; unsigned u; } x; x.f = f;
  unsigned u = x.u;
  return (unsigned short)((u + 0x7FFFu + ((u >> 16) & 1u)) >> 16);  // RNE
}
DEV unsigned f2bf2(float lo, float hi) {
  return (unsigned)f2bf(lo) | ((unsigned)f2bf(hi) << 16);
}
DEV float sigm(float x) { return 1.0f / (1.0f + __expf(-x)); }

// ---------- WMMA fragment helpers (wave32, 16x16x32 bf16) ----------
DEV v16bf load_frag16(const unsigned short* p) {   // 16 contiguous bf16 (32B)
  v16u u = *(const v16u*)p;
  return __builtin_bit_cast(v16bf, u);
}
DEV v16bf load_frag8x2(const unsigned short* p) {  // 8 @ +0 and 8 @ +16 elems
  v8us lo = *(const v8us*)p;
  v8us hi = *(const v8us*)(p + 16);
  v16u u;
#pragma unroll
  for (int j = 0; j < 8; ++j) { u[j] = lo[j]; u[8 + j] = hi[j]; }
  return __builtin_bit_cast(v16bf, u);
}
DEV v8f wmma_bf16(v16bf a, v16bf b, v8f c) {
  return __builtin_amdgcn_wmma_f32_16x16x32_bf16(false, a, false, b, (short)0, c,
                                                 false, false);
}
DEV void store_tile_bias(float* C, int N, const float* bias, int m0, int n0,
                         v8f acc) {
  int lane = threadIdx.x & 31;
  int n = n0 + (lane & 15);
  float bv = bias ? bias[n] : 0.0f;
  int mb = m0 + ((lane >> 4) << 3);
#pragma unroll
  for (int r = 0; r < 8; ++r)
    C[(long long)(mb + r) * N + n] = acc[r] + bv;
}

// ---------- one-time converters / packers ----------
__global__ void f32_to_bf16_kernel(unsigned short* dst, const float* src, int n) {
  int i = blockIdx.x * blockDim.x + threadIdx.x;
  if (i < n) dst[i] = f2bf(src[i]);
}

// trans==0: B[k,n] = src[k*ld + n]     trans==1: B[k,n] = src[n*ld + koff + k]
__global__ void pack_b_kernel(unsigned short* dst, const float* src, int K, int N,
                              int ld, int koff, int trans) {
  int i = blockIdx.x * blockDim.x + threadIdx.x;
  if (i >= K * N) return;
  int tiles_n = N >> 4;
  int tile = i >> 9;
  int lane = (i >> 4) & 31;
  int j = i & 15;
  int kt = tile / tiles_n, nt = tile % tiles_n;
  int k = (kt << 5) + ((lane >> 4) << 3) + (j & 7) + ((j >> 3) << 4);
  int n = (nt << 4) + (lane & 15);
  float v = trans ? src[(long long)n * ld + koff + k]
                  : src[(long long)k * ld + n];
  dst[i] = f2bf(v);
}

// ---------- GEMM, 16x64 per wave: C = A(bf16) @ Bpacked + bias ----------
__global__ void gemm_bf16x4_kernel(float* C, const unsigned short* A, int lda,
                                   const unsigned short* Bp, const float* bias,
                                   int M, int N, int K) {
  int wave = blockIdx.x * (blockDim.x >> 5) + (threadIdx.x >> 5);
  int tiles_n = N >> 4;
  int groups_n = tiles_n >> 2;
  int total = (M >> 4) * groups_n;
  if (wave >= total) return;              // wave-uniform
  int tm = wave / groups_n, tg = wave % groups_n;
  int m0 = tm << 4, n0 = tg << 6;
  int lane = threadIdx.x & 31;
  const unsigned short* pa =
      A + (size_t)(m0 + (lane & 15)) * lda + ((lane >> 4) << 3);
  const unsigned short* pb = Bp + (size_t)(n0 >> 4) * 512 + lane * 16;
  size_t bstep = (size_t)tiles_n * 512;
  v8f acc0 = {}, acc1 = {}, acc2 = {}, acc3 = {};
  for (int k0 = 0; k0 < K; k0 += 32) {
    v16bf a  = load_frag8x2(pa);
    v16bf b0 = load_frag16(pb);
    v16bf b1 = load_frag16(pb + 512);
    v16bf b2 = load_frag16(pb + 1024);
    v16bf b3 = load_frag16(pb + 1536);
    acc0 = wmma_bf16(a, b0, acc0);        // 4 independent WMMAs back-to-back
    acc1 = wmma_bf16(a, b1, acc1);
    acc2 = wmma_bf16(a, b2, acc2);
    acc3 = wmma_bf16(a, b3, acc3);
    pa += 32;
    pb += bstep;
  }
  store_tile_bias(C, N, bias, m0, n0,      acc0);
  store_tile_bias(C, N, bias, m0, n0 + 16, acc1);
  store_tile_bias(C, N, bias, m0, n0 + 32, acc2);
  store_tile_bias(C, N, bias, m0, n0 + 48, acc3);
}

// ---------- one-time: shifted-token embedding gather ----------
__global__ void embed_gather_kernel(unsigned short* e_bf, float* out,
                                    const long long* y, const float* embed) {
  int i = blockIdx.x * blockDim.x + threadIdx.x;
  if (i >= BT * E_) return;
  int ei = i & (E_ - 1);
  int bt = i >> 8;
  int t = bt & (T_ - 1), b = bt >> 7;
  long long tok = (t == 0) ? 0 : y[(long long)b * T_ + (t - 1)];  // PAD_IDX=0
  float v = embed[tok * E_ + ei];
  e_bf[i] = f2bf(v);
  out[(long long)bt * OUTW + (H_ + D_) + ei] = v;
}

// ---------- one-time: init h(bf16)/c(f32), combine biases ----------
__global__ void init_state_kernel(unsigned short* h_bf, float* c,
                                  const float* h0, const float* c0, float* bc,
                                  const float* b_ih, const float* b_hh) {
  int i = blockIdx.x * blockDim.x + threadIdx.x;
  if (i < B_ * H_) { h_bf[i] = f2bf(h0[i]); c[i] = c0[i]; }
  if (i < G4H) bc[i] = b_ih[i] + b_hh[i];
}

// ---------- per-step: attention MLP head ----------
// Block covers 32 rows; stages t1 = tanh(P + hp) in LDS as bf16 pairs.
// 8 waves: {m-subtile 0/1} x {n0 = 0,32,64,96}; 16x32 slab per wave.
__global__ void attn_kernel(const float* P, const float* hp,
                            const unsigned short* w2p, const float* b2,
                            const float* w3, const float* b3, float* logits) {
  __shared__ __align__(16) unsigned t1u[32 * 128];   // 32 rows x 256 bf16, 16KB
  __shared__ float lgt[32];
  const unsigned short* t1s = (const unsigned short*)t1u;
  int m0 = blockIdx.x << 5;
  for (int i = threadIdx.x; i < 32 * 128; i += blockDim.x) {
    int r = i >> 7, kp = (i & 127) << 1;
    int m = m0 + r;
    int b = m / L_;
    const float* Pm = P + (long long)m * 256;
    const float* hb = hp + b * 256;
    float x0 = tanhf(Pm[kp] + hb[kp]);
    float x1 = tanhf(Pm[kp + 1] + hb[kp + 1]);
    t1u[i] = f2bf2(x0, x1);
  }
  if (threadIdx.x < 32) lgt[threadIdx.x] = 0.0f;
  __syncthreads();

  int wv = threadIdx.x >> 5;
  int sub = wv >> 2;
  int n0 = (wv & 3) << 5;
  int lane = threadIdx.x & 31;
  int rowL = (sub << 4) + (lane & 15);
  const unsigned short* pl = t1s + rowL * 256 + ((lane >> 4) << 3);
  const unsigned short* pb = w2p + (size_t)(n0 >> 4) * 512 + lane * 16;
  v8f acc0 = {}, acc1 = {};
  for (int k0 = 0; k0 < 256; k0 += 32) {
    v16bf a  = load_frag8x2(pl);          // 2x ds_load_b128
    v16bf b0 = load_frag16(pb);
    v16bf b1 = load_frag16(pb + 512);
    acc0 = wmma_bf16(a, b0, acc0);
    acc1 = wmma_bf16(a, b1, acc1);
    pl += 32;
    pb += 8 * 512;                        // tiles_n = 8
  }
  int rbase = (sub << 4) + ((lane >> 4) << 3);
  int na = n0 + (lane & 15), nb = na + 16;
  float b2a = b2[na], b2b = b2[nb], w3a = w3[na], w3b = w3[nb];
#pragma unroll
  for (int r = 0; r < 8; ++r) {
    float s = tanhf(acc0[r] + b2a) * w3a + tanhf(acc1[r] + b2b) * w3b;
    atomicAdd(&lgt[rbase + r], s);        // ds_add_f32 reduction over n
  }
  __syncthreads();
  if (threadIdx.x < 32) logits[m0 + threadIdx.x] = lgt[threadIdx.x] + b3[0];
}

// ---------- per-step: softmax over L, z = alpha^T a ----------
__global__ void softmax_z_kernel(const float* logits, const float* a,
                                 unsigned short* z_bf, float* out, int t) {
  __shared__ float sl[256];
  __shared__ float red[256];
  int b = blockIdx.x, tid = threadIdx.x;
  float v = (tid < L_) ? logits[b * L_ + tid] : -3.4e38f;
  red[tid] = v;
  __syncthreads();
  for (int s = 128; s > 0; s >>= 1) {
    if (tid < s) red[tid] = fmaxf(red[tid], red[tid + s]);
    __syncthreads();
  }
  float mx = red[0];
  __syncthreads();
  float ev = (tid < L_) ? __expf(v - mx) : 0.0f;
  sl[tid] = ev;
  red[tid] = ev;
  __syncthreads();
  for (int s = 128; s > 0; s >>= 1) {
    if (tid < s) red[tid] += red[tid + s];
    __syncthreads();
  }
  float inv = 1.0f / red[0];
  for (int d = tid; d < D_; d += 256) {
    const float* ab = a + ((long long)b * L_) * D_ + d;
    float accv = 0.0f;
    for (int l = 0; l < L_; ++l) accv += sl[l] * ab[(long long)l * D_];
    accv *= inv;
    z_bf[b * D_ + d] = f2bf(accv);
    out[(long long)(b * T_ + t) * OUTW + H_ + d] = accv;
  }
}

// ---------- per-step: gates = z@WihZ^T + h@Whh^T + Eg[:,t,:] ----------
__global__ void gates_kernel(float* gates, const unsigned short* z_bf,
                             const unsigned short* h_bf,
                             const unsigned short* wihZp,
                             const unsigned short* whhp, const float* Eg, int t) {
  int wave = blockIdx.x * (blockDim.x >> 5) + (threadIdx.x >> 5);
  int tiles_n = G4H >> 4;                 // 128
  int groups_n = tiles_n >> 2;            // 32
  if (wave >= (B_ >> 4) * groups_n) return;
  int tm = wave / groups_n, tg = wave % groups_n;
  int m0 = tm << 4, n0 = tg << 6;
  int lane = threadIdx.x & 31;
  size_t arow = (size_t)(m0 + (lane & 15));
  int acol = (lane >> 4) << 3;
  size_t boff = (size_t)(n0 >> 4) * 512 + lane * 16;
  size_t bstep = (size_t)tiles_n * 512;
  v8f acc0 = {}, acc1 = {}, acc2 = {}, acc3 = {};
  {
    const unsigned short* pa = z_bf + arow * D_ + acol;
    const unsigned short* pb = wihZp + boff;
    for (int k0 = 0; k0 < D_; k0 += 32) {
      v16bf a  = load_frag8x2(pa);
      v16bf b0 = load_frag16(pb);
      v16bf b1 = load_frag16(pb + 512);
      v16bf b2 = load_frag16(pb + 1024);
      v16bf b3 = load_frag16(pb + 1536);
      acc0 = wmma_bf16(a, b0, acc0);
      acc1 = wmma_bf16(a, b1, acc1);
      acc2 = wmma_bf16(a, b2, acc2);
      acc3 = wmma_bf16(a, b3, acc3);
      pa += 32;
      pb += bstep;
    }
  }
  {
    const unsigned short* pa = h_bf + arow * H_ + acol;
    const unsigned short* pb = whhp + boff;
    for (int k0 = 0; k0 < H_; k0 += 32) {
      v16bf a  = load_frag8x2(pa);
      v16bf b0 = load_frag16(pb);
      v16bf b1 = load_frag16(pb + 512);
      v16bf b2 = load_frag16(pb + 1024);
      v16bf b3 = load_frag16(pb + 1536);
      acc0 = wmma_bf16(a, b0, acc0);
      acc1 = wmma_bf16(a, b1, acc1);
      acc2 = wmma_bf16(a, b2, acc2);
      acc3 = wmma_bf16(a, b3, acc3);
      pa += 32;
      pb += bstep;
    }
  }
  int mb = m0 + ((lane >> 4) << 3);
  v8f accs[4] = {acc0, acc1, acc2, acc3};
#pragma unroll
  for (int q = 0; q < 4; ++q) {
    int n = n0 + q * 16 + (lane & 15);
#pragma unroll
    for (int r = 0; r < 8; ++r) {
      int m = mb + r;
      gates[(long long)m * G4H + n] =
          accs[q][r] + Eg[(long long)(m * T_ + t) * G4H + n];
    }
  }
}

// ---------- per-step: LSTM cell elementwise ----------
__global__ void lstm_cell_kernel(const float* gates, unsigned short* h_bf,
                                 float* c, float* out, int t) {
  int i = blockIdx.x * blockDim.x + threadIdx.x;
  if (i >= B_ * H_) return;
  int b = i >> 9, j = i & (H_ - 1);
  const float* g = gates + (long long)b * G4H;
  float ig = sigm(g[j]);
  float fg = sigm(g[H_ + j]);
  float gg = tanhf(g[2 * H_ + j]);
  float og = sigm(g[3 * H_ + j]);
  float cn = fg * c[i] + ig * gg;
  float hn = og * tanhf(cn);
  c[i] = cn;
  h_bf[i] = f2bf(hn);
  out[(long long)(b * T_ + t) * OUTW + j] = hn;
}

// ================= host side =================
extern "C" void kernel_launch(void* const* d_in, const int* in_sizes, int n_in,
                              void* d_out, int out_size, void* d_ws,
                              size_t ws_size, hipStream_t stream) {
  const float*     a     = (const float*)d_in[0];
  const float*     h0    = (const float*)d_in[1];
  const float*     c0    = (const float*)d_in[2];
  const long long* y     = (const long long*)d_in[3];
  const float*     embed = (const float*)d_in[4];
  const float*     w1    = (const float*)d_in[5];
  const float*     b1    = (const float*)d_in[6];
  const float*     w2    = (const float*)d_in[7];
  const float*     b2    = (const float*)d_in[8];
  const float*     w3    = (const float*)d_in[9];
  const float*     b3    = (const float*)d_in[10];
  const float*     w_ih  = (const float*)d_in[11];
  const float*     b_ih  = (const float*)d_in[12];
  const float*     w_hh  = (const float*)d_in[13];
  const float*     b_hh  = (const float*)d_in[14];
  float* out = (float*)d_out;
  char*  ws  = (char*)d_ws;

  size_t off = 0;
  auto alloc = [&](size_t bytes) {
    size_t o = off;
    off = (off + bytes + 255) & ~(size_t)255;
    return o;
  };
  float* c      = (float*)(ws + alloc(B_ * H_ * 4));
  float* hp     = (float*)(ws + alloc(B_ * 256 * 4));
  float* logits = (float*)(ws + alloc(ML * 4));
  float* gates  = (float*)(ws + alloc(B_ * G4H * 4));
  float* bc     = (float*)(ws + alloc(G4H * 4));
  float* P      = (float*)(ws + alloc((size_t)ML * 256 * 4));
  float* Eg     = (float*)(ws + alloc((size_t)BT * G4H * 4));
  unsigned short* a_bf   = (unsigned short*)(ws + alloc((size_t)ML * D_ * 2));
  unsigned short* e_bf   = (unsigned short*)(ws + alloc((size_t)BT * E_ * 2));
  unsigned short* z_bf   = (unsigned short*)(ws + alloc(B_ * D_ * 2));
  unsigned short* h_bf   = (unsigned short*)(ws + alloc(B_ * H_ * 2));
  unsigned short* w1a_p  = (unsigned short*)(ws + alloc(512 * 256 * 2));
  unsigned short* w1h_p  = (unsigned short*)(ws + alloc(512 * 256 * 2));
  unsigned short* w2_p   = (unsigned short*)(ws + alloc(256 * 128 * 2));
  unsigned short* wihE_p = (unsigned short*)(ws + alloc(256 * 2048 * 2));
  unsigned short* wihZ_p = (unsigned short*)(ws + alloc(512 * 2048 * 2));
  unsigned short* whh_p  = (unsigned short*)(ws + alloc(512 * 2048 * 2));

  init_state_kernel<<<(B_ * H_ + 255) / 256, 256, 0, stream>>>(
      h_bf, c, h0, c0, bc, b_ih, b_hh);
  f32_to_bf16_kernel<<<(ML * D_ + 255) / 256, 256, 0, stream>>>(a_bf, a,
                                                                ML * D_);
  pack_b_kernel<<<(512 * 256 + 255) / 256, 256, 0, stream>>>(
      w1a_p, w1, 512, 256, 256, 0, 0);
  pack_b_kernel<<<(512 * 256 + 255) / 256, 256, 0, stream>>>(
      w1h_p, w1 + 512 * 256, 512, 256, 256, 0, 0);
  pack_b_kernel<<<(256 * 128 + 255) / 256, 256, 0, stream>>>(
      w2_p, w2, 256, 128, 128, 0, 0);
  pack_b_kernel<<<(256 * 2048 + 255) / 256, 256, 0, stream>>>(
      wihE_p, w_ih, 256, 2048, D_ + E_, D_, 1);
  pack_b_kernel<<<(512 * 2048 + 255) / 256, 256, 0, stream>>>(
      wihZ_p, w_ih, 512, 2048, D_ + E_, 0, 1);
  pack_b_kernel<<<(512 * 2048 + 255) / 256, 256, 0, stream>>>(
      whh_p, w_hh, 512, 2048, H_, 0, 1);
  embed_gather_kernel<<<(BT * E_ + 255) / 256, 256, 0, stream>>>(e_bf, out, y,
                                                                 embed);
  {  // P = a @ w1[:512] + b1
    int waves = (ML >> 4) * (256 >> 6);
    gemm_bf16x4_kernel<<<(waves + 3) / 4, 128, 0, stream>>>(P, a_bf, D_, w1a_p,
                                                            b1, ML, 256, D_);
  }
  {  // Eg = e @ w_ihE^T + (b_ih + b_hh)
    int waves = (BT >> 4) * (G4H >> 6);
    gemm_bf16x4_kernel<<<(waves + 3) / 4, 128, 0, stream>>>(Eg, e_bf, E_, wihE_p,
                                                            bc, BT, G4H, E_);
  }

  for (int t = 0; t < T_; ++t) {
    {  // hp = h @ w1[512:]
      int waves = (B_ >> 4) * (256 >> 6);
      gemm_bf16x4_kernel<<<(waves + 3) / 4, 128, 0, stream>>>(
          hp, h_bf, H_, w1h_p, nullptr, B_, 256, H_);
    }
    attn_kernel<<<ML / 32, 256, 0, stream>>>(P, hp, w2_p, b2, w3, b3, logits);
    softmax_z_kernel<<<B_, 256, 0, stream>>>(logits, a, z_bf, out, t);
    {
      int waves = (B_ >> 4) * (G4H >> 6);
      gates_kernel<<<(waves + 3) / 4, 128, 0, stream>>>(gates, z_bf, h_bf,
                                                        wihZ_p, whh_p, Eg, t);
    }
    lstm_cell_kernel<<<(B_ * H_ + 255) / 256, 256, 0, stream>>>(gates, h_bf, c,
                                                                out, t);
  }
}
